// MultiHeadAttentionOne_27839978012934
// MI455X (gfx1250) — matrix operand order repaired and checked by reading, not compiled, not run on hardware
//
#include <hip/hip_runtime.h>

typedef __attribute__((ext_vector_type(16))) _Float16 v16h;
typedef __attribute__((ext_vector_type(8)))  _Float16 v8h;
typedef __attribute__((ext_vector_type(8)))  float    v8f;

#define NB 8
#define LQ 1024
#define CD 512
#define NH 8
#define DKH 64

__device__ __forceinline__ int a_kmap(int e, int half) {
    return e + 8 * half + ((e >= 8) ? 8 : 0);
}

__device__ __forceinline__ v8f wmma_f16(v16h a, v16h b, v8f c) {
    return __builtin_amdgcn_wmma_f32_16x16x32_f16(
        false, a, false, b, (short)0, c, false, false);
}

// A-operand (16x32 f16): per-lane 16 elems = two contiguous 8-elem chunks at
// row+8*half and row+16+8*half (16B aligned).
__device__ __forceinline__ v16h load_a_row(const _Float16* row, int half) {
    const v8h lo = *(const v8h*)(row + 8 * half);
    const v8h hi = *(const v8h*)(row + 16 + 8 * half);
    v16h a;
#pragma unroll
    for (int e = 0; e < 8; ++e) { a[e] = lo[e]; a[e + 8] = hi[e]; }
    return a;
}

// B-operand (32x16 f16): per-lane 16 elems = two contiguous 8-elem chunks at
// 8*half and 16+8*half (same K order as A).
__device__ __forceinline__ v16h load_b_row(const _Float16* colbase, int half) {
    const v8h lo = *(const v8h*)(colbase + 8 * half);
    const v8h hi = *(const v8h*)(colbase + 16 + 8 * half);
    v16h b;
#pragma unroll
    for (int e = 0; e < 8; ++e) { b[e] = lo[e]; b[e + 8] = hi[e]; }
    return b;
}

// ---------------------------------------------------------------------------
// Kernel 1: QKV projection. One wave -> 16 tokens x 64 out-channels (= 1 head).
// A tile (x, transpose-gathered from NCHW q) loaded once per K-step, reused 4x.
// ---------------------------------------------------------------------------
__global__ void qkv_proj_kernel(const float* __restrict__ q,
                                const float* __restrict__ w,
                                _Float16* __restrict__ qh) {
    const int lane = threadIdx.x;
    const int half = lane >> 4;
    const int nidx = lane & 15;
    const int nTilesN = CD / 64;              // 8 (= heads)
    const int tileN = blockIdx.x % nTilesN;
    const int tileM = blockIdx.x / nTilesN;
    const int tok0  = tileM * 16;
    const int b     = tok0 / LQ;
    const int l0    = tok0 % LQ;
    const int n0    = tileN * 64;             // head tileN

    v8f acc[4] = {};
    for (int kc = 0; kc < CD; kc += 32) {
        v16h a;
#pragma unroll
        for (int e = 0; e < 16; ++e) {
            int c = kc + 16 * half + e;
            a[e] = (_Float16)q[((size_t)b * CD + c) * LQ + l0 + nidx];
        }
#pragma unroll
        for (int t = 0; t < 4; ++t) {
            const float* wr = w + (size_t)(n0 + t * 16 + nidx) * CD + kc + 16 * half;
            v16h bb;
#pragma unroll
            for (int e = 0; e < 16; ++e) bb[e] = (_Float16)wr[e];
            acc[t] = wmma_f16(a, bb, acc[t]);
        }
    }
#pragma unroll
    for (int t = 0; t < 4; ++t)
#pragma unroll
        for (int r = 0; r < 8; ++r) {
            int l = l0 + r + 8 * half;
            qh[(((size_t)b * NH + tileN) * LQ + l) * DKH + t * 16 + nidx] =
                (_Float16)acc[t][r];
        }
}

// ---------------------------------------------------------------------------
// Kernel 2: fused flash attention. 4 waves (64 queries) per block share the
// 32-key K/V tile through LDS (row-major for scores-B, transposed for PV-B).
// ---------------------------------------------------------------------------
__global__ void attn_kernel(const _Float16* __restrict__ qh,
                            _Float16* __restrict__ ov) {
    __shared__ __align__(16) _Float16 Kv[32 * DKH];      // K rows, row-major
    __shared__ __align__(16) _Float16 Vt[DKH * 32];      // V transposed [d][key]
    __shared__ __align__(16) _Float16 Plds[4 * 16 * 32]; // per-wave P tile
    const int tid  = threadIdx.x;
    const int wave = tid >> 5;
    const int lane = tid & 31;
    const int half = lane >> 4;
    const int nidx = lane & 15;
    const int bh = blockIdx.x >> 4;           // 16 query-groups per (b,h)
    const int qt = blockIdx.x & 15;
    const int b = bh / NH, h = bh % NH;
    const int q0 = qt * 64 + wave * 16;
    const _Float16* Q = qh + (size_t)(b * NH + h) * LQ * DKH;
    _Float16* P = Plds + wave * 16 * 32;

    const v16h aq0 = load_a_row(Q + (size_t)(q0 + nidx) * DKH, half);
    const v16h aq1 = load_a_row(Q + (size_t)(q0 + nidx) * DKH + 32, half);

    float rmax[8], rsum[8];
#pragma unroll
    for (int r = 0; r < 8; ++r) { rmax[r] = -1e30f; rsum[r] = 0.0f; }
    v8f acc[4] = {};
    const float scale = 0.125f;

    for (int kb = 0; kb < LQ; kb += 32) {
        __syncthreads();                      // all waves done with prev tile
        {
            // cooperative stage: 32 keys x 64 d = 4KB, contiguous in qh
            const int key = tid >> 2;
            const int d0  = (tid & 3) * 16;
            const v8h s0 = *(const v8h*)(Q + (size_t)(kb + key) * DKH + d0);
            const v8h s1 = *(const v8h*)(Q + (size_t)(kb + key) * DKH + d0 + 8);
            *(v8h*)(Kv + key * DKH + d0)     = s0;
            *(v8h*)(Kv + key * DKH + d0 + 8) = s1;
#pragma unroll
            for (int j = 0; j < 8; ++j) {
                Vt[(d0 + j) * 32 + key]     = s0[j];
                Vt[(d0 + 8 + j) * 32 + key] = s1[j];
            }
        }
        if (kb + 32 < LQ)
            __builtin_prefetch(Q + (size_t)(kb + 32) * DKH + (tid & 31) * 64, 0, 1);
        __syncthreads();                      // tile visible to all waves

        v8f st[2];
#pragma unroll
        for (int s = 0; s < 2; ++s) {
            const _Float16* krow = Kv + (s * 16 + nidx) * DKH;
            v8f sa = {};
            sa = wmma_f16(aq0, load_b_row(krow, half), sa);
            sa = wmma_f16(aq1, load_b_row(krow + 32, half), sa);
#pragma unroll
            for (int r = 0; r < 8; ++r) sa[r] *= scale;
            st[s] = sa;
        }

        float alpha[8];
#pragma unroll
        for (int r = 0; r < 8; ++r) {
            float mloc = fmaxf(st[0][r], st[1][r]);
#pragma unroll
            for (int m = 1; m <= 8; m <<= 1)
                mloc = fmaxf(mloc, __shfl_xor(mloc, m, 32));
            float mnew = fmaxf(rmax[r], mloc);
            float a  = __expf(rmax[r] - mnew);
            float p0 = __expf(st[0][r] - mnew);
            float p1 = __expf(st[1][r] - mnew);
            float ps = p0 + p1;
#pragma unroll
            for (int m = 1; m <= 8; m <<= 1)
                ps += __shfl_xor(ps, m, 32);
            rsum[r] = rsum[r] * a + ps;
            rmax[r] = mnew;
            alpha[r] = a;
            st[0][r] = p0; st[1][r] = p1;
        }

        // D-layout -> A-layout relayout of P (per-wave LDS region)
#pragma unroll
        for (int r = 0; r < 8; ++r) {
            int mrow = r + 8 * half;
            P[mrow * 32 + nidx]      = (_Float16)st[0][r];
            P[mrow * 32 + 16 + nidx] = (_Float16)st[1][r];
        }
#pragma unroll
        for (int t = 0; t < 4; ++t)
#pragma unroll
            for (int r = 0; r < 8; ++r) acc[t][r] *= alpha[r];

        const v16h ap = load_a_row(P + nidx * 32, half);
#pragma unroll
        for (int t = 0; t < 4; ++t) {
            const v16h bv = load_b_row(Vt + (t * 16 + nidx) * 32, half);
            acc[t] = wmma_f16(ap, bv, acc[t]);
        }
    }

#pragma unroll
    for (int r = 0; r < 8; ++r) rsum[r] = 1.0f / rsum[r];
#pragma unroll
    for (int t = 0; t < 4; ++t)
#pragma unroll
        for (int r = 0; r < 8; ++r) {
            int l = q0 + r + 8 * half;
            ov[((size_t)b * LQ + l) * CD + h * DKH + t * 16 + nidx] =
                (_Float16)(acc[t][r] * rsum[r]);
        }
}

// ---------------------------------------------------------------------------
// Kernel 3: FC + bias + residual. One wave -> 16 tokens x 64 channels.
// ---------------------------------------------------------------------------
__global__ void fc_kernel(const _Float16* __restrict__ ov,
                          const float* __restrict__ fcw,
                          const float* __restrict__ fcb,
                          const float* __restrict__ q,
                          float* __restrict__ y) {
    const int lane = threadIdx.x;
    const int half = lane >> 4;
    const int nidx = lane & 15;
    const int nTilesN = CD / 64;
    const int tileN = blockIdx.x % nTilesN;
    const int tileM = blockIdx.x / nTilesN;
    const int tok0 = tileM * 16;
    const int n0 = tileN * 64;

    v8f acc[4] = {};
    for (int kc = 0; kc < CD; kc += 32) {
        const v16h a = load_a_row(ov + (size_t)(tok0 + nidx) * CD + kc, half);
#pragma unroll
        for (int t = 0; t < 4; ++t) {
            const float* wr = fcw + (size_t)(n0 + t * 16 + nidx) * CD + kc;
            v16h bb;
#pragma unroll
            for (int e = 0; e < 16; ++e) bb[e] = (_Float16)wr[a_kmap(e, half)];
            acc[t] = wmma_f16(a, bb, acc[t]);
        }
    }
#pragma unroll
    for (int t = 0; t < 4; ++t) {
        const int j = n0 + t * 16 + nidx;
        const float bias = fcb[j];
#pragma unroll
        for (int r = 0; r < 8; ++r) {
            int tok = tok0 + r + 8 * half;
            int b = tok / LQ, l = tok % LQ;
            float res = q[((size_t)b * CD + j) * LQ + l];
            y[(size_t)tok * CD + j] = acc[t][r] + bias + res;
        }
    }
}

// ---------------------------------------------------------------------------
// Kernel 4: LayerNorm over 512 channels, in place on d_out.
// ---------------------------------------------------------------------------
__global__ void ln_kernel(float* __restrict__ y,
                          const float* __restrict__ gamma,
                          const float* __restrict__ beta) {
    __shared__ float red[256];
    const int tok = blockIdx.x;
    const int t = threadIdx.x;
    float v0 = y[(size_t)tok * CD + t];
    float v1 = y[(size_t)tok * CD + t + 256];
    red[t] = v0 + v1;
    __syncthreads();
    for (int o = 128; o > 0; o >>= 1) {
        if (t < o) red[t] += red[t + o];
        __syncthreads();
    }
    float mu = red[0] * (1.0f / CD);
    __syncthreads();
    float d0 = v0 - mu, d1 = v1 - mu;
    red[t] = d0 * d0 + d1 * d1;
    __syncthreads();
    for (int o = 128; o > 0; o >>= 1) {
        if (t < o) red[t] += red[t + o];
        __syncthreads();
    }
    float rstd = rsqrtf(red[0] * (1.0f / CD) + 1e-5f);
    y[(size_t)tok * CD + t]       = d0 * rstd * gamma[t] + beta[t];
    y[(size_t)tok * CD + t + 256] = d1 * rstd * gamma[t + 256] + beta[t + 256];
}

extern "C" void kernel_launch(void* const* d_in, const int* in_sizes, int n_in,
                              void* d_out, int out_size, void* d_ws, size_t ws_size,
                              hipStream_t stream) {
    const float* q    = (const float*)d_in[0];
    const float* wqkv = (const float*)d_in[1];
    const float* fcw  = (const float*)d_in[2];
    const float* fcb  = (const float*)d_in[3];
    const float* gam  = (const float*)d_in[4];
    const float* bet  = (const float*)d_in[5];
    float* y = (float*)d_out;

    _Float16* qh = (_Float16*)d_ws;
    _Float16* ov = qh + (size_t)NB * NH * LQ * DKH;

    dim3 wave(32);
    qkv_proj_kernel<<<dim3((NB * LQ / 16) * (CD / 64)), wave, 0, stream>>>(q, wqkv, qh);
    attn_kernel<<<dim3(NB * NH * (LQ / 64)), dim3(128), 0, stream>>>(qh, ov);
    fc_kernel<<<dim3((NB * LQ / 16) * (CD / 64)), wave, 0, stream>>>(ov, fcw, fcb, q, y);
    ln_kernel<<<dim3(NB * LQ), dim3(256), 0, stream>>>(y, gam, bet);
}
